// ScaledMultiheadAttention_84817014161532
// MI455X (gfx1250) — matrix operand order repaired and checked by reading, not compiled, not run on hardware
//
#include <hip/hip_runtime.h>

// ---------------- problem constants ----------------
constexpr int S_ = 1024;   // sequence length
constexpr int B_ = 4;      // batch
constexpr int E_ = 1024;   // embed dim
constexpr int H_ = 16;     // heads
constexpr int D_ = 64;     // head dim
constexpr int R_ = S_ * B_;  // 4096 rows of the [S*B, E] activations

// ---------------- WMMA types (CDNA5 / gfx1250, wave32) ----------------
typedef __bf16 v16bf __attribute__((ext_vector_type(16)));
typedef __bf16 v8bf  __attribute__((ext_vector_type(8)));
typedef float  v8f   __attribute__((ext_vector_type(8)));

static __device__ __forceinline__ v8f v8f_zero() {
    v8f z;
#pragma unroll
    for (int i = 0; i < 8; ++i) z[i] = 0.0f;
    return z;
}

// f32 -> bf16, round to nearest even
static __device__ __forceinline__ unsigned short f2bf(float f) {
    unsigned u = __builtin_bit_cast(unsigned, f);
    u += 0x7FFFu + ((u >> 16) & 1u);
    return (unsigned short)(u >> 16);
}

// D = A(16x32 bf16) * B(32x16 bf16) + C(16x16 f32)
static __device__ __forceinline__ v8f wmma_bf16(v16bf a, v16bf b, v8f c) {
    return __builtin_amdgcn_wmma_f32_16x16x32_bf16(false, a, false, b,
                                                   (short)0, c, false, false);
}

// A-fragment (16x32, 16-bit): lane m = lane&15, hi = lane>>4.
// Element i -> K = kb + hi*16 + i (contiguous order).
// Row-major source with contiguous K => two 16-byte runs.
static __device__ __forceinline__ v16bf make_afrag(const __bf16* row, int kb, int hi) {
    v8bf r0 = *(const v8bf*)(row + kb + hi * 16);
    v8bf r1 = *(const v8bf*)(row + kb + hi * 16 + 8);
    v16bf a;
#pragma unroll
    for (int i = 0; i < 8; ++i) { a[i] = r0[i]; a[i + 8] = r1[i]; }
    return a;
}

// lane-XOR shuffle via ds_swizzle (group-of-32: offset = xor<<10 | and_mask 0x1f)
template <int XORM>
static __device__ __forceinline__ float swz_xor(float v) {
    int r = __builtin_amdgcn_ds_swizzle(__builtin_bit_cast(int, v),
                                        (XORM << 10) | 0x1f);
    return __builtin_bit_cast(float, r);
}
static __device__ __forceinline__ float redmax16(float v) {
    v = fmaxf(v, swz_xor<1>(v)); v = fmaxf(v, swz_xor<2>(v));
    v = fmaxf(v, swz_xor<4>(v)); v = fmaxf(v, swz_xor<8>(v));
    return v;
}
static __device__ __forceinline__ float redsum16(float v) {
    v += swz_xor<1>(v); v += swz_xor<2>(v);
    v += swz_xor<4>(v); v += swz_xor<8>(v);
    return v;
}

// ---------------- stage 0: f32 -> bf16 conversion ----------------
__global__ __launch_bounds__(256) void cvt4(const float* __restrict__ src,
                                            unsigned short* __restrict__ dst,
                                            int n4) {
    int i = blockIdx.x * blockDim.x + threadIdx.x;
    if (i >= n4) return;
    const float4 v = ((const float4*)src)[i];
    unsigned long long p =
        (unsigned long long)f2bf(v.x) |
        ((unsigned long long)f2bf(v.y) << 16) |
        ((unsigned long long)f2bf(v.z) << 32) |
        ((unsigned long long)f2bf(v.w) << 48);
    ((unsigned long long*)dst)[i] = p;
}

// Software-pipelined 16x64 GEMM core: next step's fragments are loaded
// before the current step's WMMAs so loads overlap matrix math
// (partial s_wait_loadcnt instead of 0).
static __device__ __forceinline__ void gemm_16x64(const __bf16* __restrict__ arow,
                                                  const __bf16* __restrict__ brow0,
                                                  int hi, v8f acc[4]) {
    v16bf a = make_afrag(arow, 0, hi);
    v16bf b0 = *(const v16bf*)(brow0 + (size_t)(0 * 16) * E_ + hi * 16);
    v16bf b1 = *(const v16bf*)(brow0 + (size_t)(1 * 16) * E_ + hi * 16);
    v16bf b2 = *(const v16bf*)(brow0 + (size_t)(2 * 16) * E_ + hi * 16);
    v16bf b3 = *(const v16bf*)(brow0 + (size_t)(3 * 16) * E_ + hi * 16);
#pragma unroll 1
    for (int kb = 0; kb < E_ - 32; kb += 32) {
        int kn = kb + 32;
        v16bf an  = make_afrag(arow, kn, hi);
        v16bf bn0 = *(const v16bf*)(brow0 + (size_t)(0 * 16) * E_ + kn + hi * 16);
        v16bf bn1 = *(const v16bf*)(brow0 + (size_t)(1 * 16) * E_ + kn + hi * 16);
        v16bf bn2 = *(const v16bf*)(brow0 + (size_t)(2 * 16) * E_ + kn + hi * 16);
        v16bf bn3 = *(const v16bf*)(brow0 + (size_t)(3 * 16) * E_ + kn + hi * 16);
        acc[0] = wmma_bf16(a, b0, acc[0]);
        acc[1] = wmma_bf16(a, b1, acc[1]);
        acc[2] = wmma_bf16(a, b2, acc[2]);
        acc[3] = wmma_bf16(a, b3, acc[3]);
        a = an; b0 = bn0; b1 = bn1; b2 = bn2; b3 = bn3;
    }
    acc[0] = wmma_bf16(a, b0, acc[0]);
    acc[1] = wmma_bf16(a, b1, acc[1]);
    acc[2] = wmma_bf16(a, b2, acc[2]);
    acc[3] = wmma_bf16(a, b3, acc[3]);
}

// ---------------- stage 1: fused QKV projection ----------------
// Each wave computes a 16x64 output tile (4 WMMA accumulators sharing one
// A-fragment per K-step). The 64 columns of a wave tile are exactly one
// head: h = nt4, d = j*16+ln.
// Y = X @ W^T + b ; Q *= 1/sqrt(D) ; V *= learned_scale[h] ; relayout:
//   Qb, Kb : [B,H,S,D] bf16      Vt : [B,H,D,S] bf16 (transposed for P@V)
__global__ __launch_bounds__(256) void qkv_proj(const unsigned short* __restrict__ Xus,
                                                const unsigned short* __restrict__ Wus,
                                                const float* __restrict__ bias,
                                                const float* __restrict__ lscale,
                                                unsigned short* __restrict__ Qb,
                                                unsigned short* __restrict__ Kb,
                                                unsigned short* __restrict__ Vt) {
    const __bf16* Xb = (const __bf16*)Xus;
    const __bf16* Wb = (const __bf16*)Wus;
    int wave = threadIdx.x >> 5;
    int lane = threadIdx.x & 31;
    int ln = lane & 15, hi = lane >> 4;
    int t = blockIdx.x * 8 + wave;        // 0 .. 3*256*16-1
    int proj = t >> 12;                   // 0=q 1=k 2=v
    int t2 = t & 4095;
    int mt = t2 >> 4, nt4 = t2 & 15;      // 16-row tile, 64-col tile

    const __bf16* arow = Xb + (size_t)proj * R_ * E_ + (size_t)(mt * 16 + ln) * E_;
    const __bf16* brow0 = Wb + (size_t)proj * E_ * E_ + (size_t)(nt4 * 64 + ln) * E_;

    v8f acc[4];
#pragma unroll
    for (int j = 0; j < 4; ++j) acc[j] = v8f_zero();

    gemm_16x64(arow, brow0, hi, acc);

    int h = nt4;                       // this wave's 64 cols = one head
    float vscale = lscale[h];

#pragma unroll
    for (int j = 0; j < 4; ++j) {
        int d = j * 16 + ln;
        int ncol = nt4 * 64 + d;
        float bv = bias[proj * E_ + ncol];
        if (proj == 0) {
#pragma unroll
            for (int r = 0; r < 8; ++r) {
                int rg = mt * 16 + r + hi * 8;
                int s = rg >> 2, bb = rg & 3;
                float v = (acc[j][r] + bv) * 0.125f;  // 1/sqrt(64)
                Qb[(((size_t)(bb * H_ + h)) * S_ + s) * D_ + d] = f2bf(v);
            }
        } else if (proj == 1) {
#pragma unroll
            for (int r = 0; r < 8; ++r) {
                int rg = mt * 16 + r + hi * 8;
                int s = rg >> 2, bb = rg & 3;
                Kb[(((size_t)(bb * H_ + h)) * S_ + s) * D_ + d] = f2bf(acc[j][r] + bv);
            }
        } else {
#pragma unroll
            for (int r = 0; r < 8; ++r) {
                int rg = mt * 16 + r + hi * 8;
                int s = rg >> 2, bb = rg & 3;
                Vt[(((size_t)(bb * H_ + h)) * D_ + d) * S_ + s] =
                    f2bf((acc[j][r] + bv) * vscale);
            }
        }
    }
}

// ---------------- stage 2: flash attention per (b,h) ----------------
// block = 4 waves, each wave owns a 16-query tile; 64-key steps with
// online softmax; P staged through LDS (C-layout -> A-fragment layout).
__global__ __launch_bounds__(128) void flash_attn(const unsigned short* __restrict__ Qus,
                                                  const unsigned short* __restrict__ Kus,
                                                  const unsigned short* __restrict__ Vus,
                                                  unsigned short* __restrict__ X2) {
    __shared__ unsigned short P[4 * 16 * 64];

    const __bf16* Qb = (const __bf16*)Qus;
    const __bf16* Kb = (const __bf16*)Kus;
    const __bf16* Vt = (const __bf16*)Vus;

    int wave = threadIdx.x >> 5;
    int lane = threadIdx.x & 31;
    int ln = lane & 15, hi = lane >> 4;
    int bh = blockIdx.x >> 4;   // b*H + h
    int qb = blockIdx.x & 15;
    int qbase = qb * 64 + wave * 16;

    const __bf16* qrow = Qb + ((size_t)bh * S_ + qbase + ln) * D_;
    v16bf Aq0 = make_afrag(qrow, 0, hi);
    v16bf Aq1 = make_afrag(qrow, 32, hi);

    v8f oacc[4];
#pragma unroll
    for (int dt = 0; dt < 4; ++dt) oacc[dt] = v8f_zero();
    float mx[8], lsum[8];
#pragma unroll
    for (int r = 0; r < 8; ++r) { mx[r] = -3.0e38f; lsum[r] = 0.0f; }

    unsigned short* Pw = &P[wave * 16 * 64];

    for (int kb = 0; kb < S_; kb += 64) {
        // ---- scores: S(16x64) = Q(16x64d) @ K^T ----
        v8f sc[4];
#pragma unroll
        for (int jn = 0; jn < 4; ++jn) {
            const __bf16* kr = Kb + ((size_t)bh * S_ + kb + jn * 16 + ln) * D_;
            v16bf b0 = *(const v16bf*)(kr + hi * 16);
            v16bf b1 = *(const v16bf*)(kr + 32 + hi * 16);
            v8f t = v8f_zero();
            t = wmma_bf16(Aq0, b0, t);
            t = wmma_bf16(Aq1, b1, t);
            sc[jn] = t;
        }
        // ---- online softmax, P -> LDS as bf16 ----
#pragma unroll
        for (int r = 0; r < 8; ++r) {
            float tm = fmaxf(fmaxf(sc[0][r], sc[1][r]), fmaxf(sc[2][r], sc[3][r]));
            tm = redmax16(tm);
            float nm = fmaxf(mx[r], tm);
            float corr = __expf(mx[r] - nm);
            mx[r] = nm;
            float p0 = __expf(sc[0][r] - nm);
            float p1 = __expf(sc[1][r] - nm);
            float p2 = __expf(sc[2][r] - nm);
            float p3 = __expf(sc[3][r] - nm);
            float rs = redsum16(p0 + p1 + p2 + p3);
            lsum[r] = lsum[r] * corr + rs;
#pragma unroll
            for (int dt = 0; dt < 4; ++dt) oacc[dt][r] *= corr;
            int rowoff = (r + hi * 8) * 64 + ln;
            Pw[rowoff +  0] = f2bf(p0);
            Pw[rowoff + 16] = f2bf(p1);
            Pw[rowoff + 32] = f2bf(p2);
            Pw[rowoff + 48] = f2bf(p3);
        }
        __syncthreads();  // DS is in-order per wave; barrier is belt-and-braces

        // ---- O += P(16x64) @ V(64x64) ----
        const __bf16* prow = (const __bf16*)&Pw[ln * 64];
        v16bf Ap0 = make_afrag(prow, 0, hi);
        v16bf Ap1 = make_afrag(prow, 32, hi);
#pragma unroll
        for (int dt = 0; dt < 4; ++dt) {
            const __bf16* vr = Vt + ((size_t)bh * D_ + dt * 16 + ln) * S_ + kb;
            v16bf b0 = *(const v16bf*)(vr + hi * 16);
            v16bf b1 = *(const v16bf*)(vr + 32 + hi * 16);
            oacc[dt] = wmma_bf16(Ap0, b0, oacc[dt]);
            oacc[dt] = wmma_bf16(Ap1, b1, oacc[dt]);
        }
        __syncthreads();
    }

    // ---- epilogue: normalize by row sum, write [S*B, E] bf16 ----
    int b = bh >> 4, h = bh & 15;
#pragma unroll
    for (int r = 0; r < 8; ++r) {
        float inv = 1.0f / lsum[r];
        int q = qbase + r + hi * 8;                    // s index
        size_t rowbase = ((size_t)q * B_ + b) * E_ + h * D_;
#pragma unroll
        for (int dt = 0; dt < 4; ++dt)
            X2[rowbase + dt * 16 + ln] = f2bf(oacc[dt][r] * inv);
    }
}

// ---------------- stage 3: output projection ----------------
// Same software-pipelined 16x64-per-wave tiling; f32 output + bias.
__global__ __launch_bounds__(256) void out_proj(const unsigned short* __restrict__ Xus,
                                                const unsigned short* __restrict__ Wus,
                                                const float* __restrict__ bias,
                                                float* __restrict__ out) {
    const __bf16* Xb = (const __bf16*)Xus;
    const __bf16* Wb = (const __bf16*)Wus;
    int wave = threadIdx.x >> 5;
    int lane = threadIdx.x & 31;
    int ln = lane & 15, hi = lane >> 4;
    int t = blockIdx.x * 8 + wave;   // 0 .. 256*16-1
    int mt = t >> 4, nt4 = t & 15;

    const __bf16* arow = Xb + (size_t)(mt * 16 + ln) * E_;
    const __bf16* brow0 = Wb + (size_t)(nt4 * 64 + ln) * E_;

    v8f acc[4];
#pragma unroll
    for (int j = 0; j < 4; ++j) acc[j] = v8f_zero();

    gemm_16x64(arow, brow0, hi, acc);

#pragma unroll
    for (int j = 0; j < 4; ++j) {
        int ncol = nt4 * 64 + j * 16 + ln;
        float bv = bias[ncol];
#pragma unroll
        for (int r = 0; r < 8; ++r) {
            int rg = mt * 16 + r + hi * 8;
            out[(size_t)rg * E_ + ncol] = acc[j][r] + bv;
        }
    }
}

// ---------------- host launcher ----------------
extern "C" void kernel_launch(void* const* d_in, const int* in_sizes, int n_in,
                              void* d_out, int out_size, void* d_ws, size_t ws_size,
                              hipStream_t stream) {
    const float* q_f   = (const float*)d_in[0];
    const float* k_f   = (const float*)d_in[1];
    const float* v_f   = (const float*)d_in[2];
    const float* win_f = (const float*)d_in[3];
    const float* bin_f = (const float*)d_in[4];
    const float* wo_f  = (const float*)d_in[5];
    const float* bo_f  = (const float*)d_in[6];
    const float* ls_f  = (const float*)d_in[7];

    // workspace layout (bytes), 64 MB total
    char* w = (char*)d_ws;
    unsigned short* XB  = (unsigned short*)(w + 0);           // 3 x [R,E] bf16  (q,k,v inputs)
    unsigned short* WIN = (unsigned short*)(w + 25165824);    // 3 x [E,E] bf16
    unsigned short* WO  = (unsigned short*)(w + 31457280);    // [E,E] bf16
    unsigned short* QB  = (unsigned short*)(w + 33554432);    // [B,H,S,D] bf16
    unsigned short* KB  = (unsigned short*)(w + 41943040);    // [B,H,S,D] bf16
    unsigned short* VT  = (unsigned short*)(w + 50331648);    // [B,H,D,S] bf16
    unsigned short* X2  = (unsigned short*)(w + 58720256);    // [R,E] bf16 (attn out)

    const int nAct = R_ * E_;          // 4194304 per tensor
    const int nWin = 3 * E_ * E_;      // 3145728
    const int nWo  = E_ * E_;          // 1048576

    cvt4<<<nAct / 1024, 256, 0, stream>>>(q_f,   XB,               nAct / 4);
    cvt4<<<nAct / 1024, 256, 0, stream>>>(k_f,   XB + nAct,        nAct / 4);
    cvt4<<<nAct / 1024, 256, 0, stream>>>(v_f,   XB + 2 * nAct,    nAct / 4);
    cvt4<<<nWin / 1024, 256, 0, stream>>>(win_f, WIN,              nWin / 4);
    cvt4<<<nWo  / 1024, 256, 0, stream>>>(wo_f,  WO,               nWo  / 4);

    // 3 projections * 256 m-tiles * 16 n64-tiles = 12288 wave-tiles / 8 waves
    qkv_proj<<<1536, 256, 0, stream>>>(XB, WIN, bin_f, ls_f, QB, KB, VT);

    // B*H * (S/64) = 64 * 16 = 1024 blocks, 4 waves each
    flash_attn<<<1024, 128, 0, stream>>>(QB, KB, VT, X2);

    // 256 * 16 = 4096 wave-tiles / 8 waves
    out_proj<<<512, 256, 0, stream>>>(X2, WO, bo_f, (float*)d_out);
}
